// NeuroNet_231928234454
// MI455X (gfx1250) — matrix operand order repaired and checked by reading, not compiled, not run on hardware
//
#include <hip/hip_runtime.h>
#include <math.h>

typedef __attribute__((ext_vector_type(16))) _Float16 v16h;
typedef __attribute__((ext_vector_type(8)))  float    v8f;
typedef __attribute__((ext_vector_type(4)))  float    v4f;

#define S_SYM  4096
#define D_DIM  128
#define DD_DIM 64
#define REL_N  8
#define R_N    512
#define N_COLS (S_SYM * REL_N)          // 32768
#define ROWLEN (REL_N * S_SYM + R_N)    // 33280
#define FC_IN  (DD_DIM * ROWLEN)        // 2,129,920
#define CHUNK  4096
#define NBLK2  (FC_IN / CHUNK)          // 520 (exact)

// ---------------------------------------------------------------------------
// Kernel 1: kg[m, n] = (dise_w[m,:] . x[n,:]) / max(|dise_w[m]| * |x[n]|, eps)
// x[n,:] = (embed_w[n/8,:] + rel_w[n%8,:]) * (inputs[n/8] != 0)
// One wave computes a 16-wide column tile against all 64 dise rows using
// v_wmma_f32_16x16x32_f16 (4 m-tiles x 4 K-steps = 16 WMMAs per wave).
// ---------------------------------------------------------------------------
__global__ __launch_bounds__(128)
void kg_kernel(const int*   __restrict__ inputs,
               const float* __restrict__ embed_w,
               const float* __restrict__ rel_w,
               const float* __restrict__ dise_w,
               float*       __restrict__ kg)
{
    __shared__ float dn_s[DD_DIM];
    const int t = threadIdx.x;

    // disease-embedding norms (64 of them), computed cooperatively
    if (t < DD_DIM) {
        const float* row = dise_w + t * D_DIM;
        float s = 0.f;
        #pragma unroll 8
        for (int d = 0; d < D_DIM; ++d) { float v = row[d]; s += v * v; }
        dn_s[t] = sqrtf(s);
    }
    __syncthreads();

    const int  lane  = t & 31;
    const int  wave  = t >> 5;
    const int  tileN = blockIdx.x * 4 + wave;   // 2048 tiles total
    const int  n0    = tileN * 16;
    const int  ln    = lane & 15;
    const bool hi    = lane >= 16;

    // This lane contributes column n of the B (x^T) fragment.
    const int   n     = n0 + ln;
    const int   s_idx = n >> 3;
    const int   rel   = n & 7;
    const float maskv = (inputs[s_idx] != 0) ? 1.0f : 0.0f;
    const float* erow = embed_w + (size_t)s_idx * D_DIM;
    const float* rrow = rel_w   + (size_t)rel   * D_DIM;

    v8f acc[4];
    #pragma unroll
    for (int mt = 0; mt < 4; ++mt) { v8f z = {}; acc[mt] = z; }

    float sq = 0.f;   // partial |x[n]|^2 (this half-wave's K ranges)

    #pragma unroll
    for (int kkStep = 0; kkStep < 4; ++kkStep) {
        const int kk = kkStep * 32;

        // B fragment (32x16, f16): lanes 0-15 -> K = kk+0..15,
        //                          lanes 16-31 -> K = kk+16..31, col = n.
        const int kb = kk + (hi ? 16 : 0);
        v16h bfrag;
        #pragma unroll
        for (int i = 0; i < 16; ++i) {
            float xv = (erow[kb + i] + rrow[kb + i]) * maskv;
            sq += xv * xv;
            bfrag[i] = (_Float16)xv;
        }

        // A fragments (16x32, f16) per 16-row dise tile:
        // lane row M = mt*16 + ln; halves 0-15 -> K = kk+koff+0..15,
        // koff = hi ? 16 : 0.
        const int koff = hi ? 16 : 0;
        #pragma unroll
        for (int mt = 0; mt < 4; ++mt) {
            const float* arow = dise_w + (size_t)(mt * 16 + ln) * D_DIM + kk + koff;
            v16h afrag;
            #pragma unroll
            for (int i = 0; i < 8; ++i) {
                afrag[i]     = (_Float16)arow[i];
                afrag[i + 8] = (_Float16)arow[8 + i];
            }
            acc[mt] = __builtin_amdgcn_wmma_f32_16x16x32_f16(
                false, afrag, false, bfrag, (short)0, acc[mt], false, false);
        }
    }

    // combine the two half-wave partial squared norms -> full |x[n]|^2
    sq += __shfl_xor(sq, 16, 32);
    const float xn = (sq > 0.f) ? sqrtf(sq) : 0.f;

    // C layout: VGPR v, lanes 0-15 -> M = v, lanes 16-31 -> M = v+8; N = ln.
    #pragma unroll
    for (int mt = 0; mt < 4; ++mt) {
        #pragma unroll
        for (int v = 0; v < 8; ++v) {
            const int m = mt * 16 + v + (hi ? 8 : 0);
            const float denom = fmaxf(dn_s[m] * xn, 1e-6f);
            kg[(size_t)m * N_COLS + n] = acc[mt][v] / denom;
        }
    }
}

// ---------------------------------------------------------------------------
// Kernel 2: streaming GEMV partials. Each block stages a 4096-float chunk of
// `hidden` in LDS once, then 8 waves stream the matching 16KB slice of all
// 64 fc_w rows with 128-bit non-temporal loads (fc_w = 545MB >> L2).
// partial[block][j] written without atomics -> deterministic.
// ---------------------------------------------------------------------------
__global__ __launch_bounds__(256)
void gemv_kernel(const float* __restrict__ rf,
                 const float* __restrict__ kg,
                 const float* __restrict__ fc_w,
                 float*       __restrict__ partial)
{
    __shared__ __align__(16) float h_lds[CHUNK];
    const unsigned base = blockIdx.x * CHUNK;

    // splice hidden = [rule_features | kg] per 33280-wide row k
    for (int i = threadIdx.x; i < CHUNK; i += 256) {
        const unsigned f = base + i;
        const unsigned k = f / ROWLEN;
        const unsigned c = f - k * ROWLEN;
        h_lds[i] = (c < R_N) ? rf[k * R_N + c]
                             : kg[(size_t)k * N_COLS + (c - R_N)];
    }
    __syncthreads();

    const int lane = threadIdx.x & 31;
    const int wave = threadIdx.x >> 5;
    const v4f* hvec = (const v4f*)h_lds;

    #pragma unroll 1
    for (int jj = 0; jj < 8; ++jj) {
        const int j = wave + jj * 8;          // all 64 rows covered per block
        const v4f* wrow = (const v4f*)(fc_w + (size_t)j * FC_IN + base);
        float a = 0.f;
        #pragma unroll 4
        for (int it = 0; it < (CHUNK / 4) / 32; ++it) {
            const int tq = it * 32 + lane;    // coalesced 512B per wave-load
            v4f w = __builtin_nontemporal_load(&wrow[tq]);
            v4f h = hvec[tq];
            a += w.x * h.x + w.y * h.y + w.z * h.z + w.w * h.w;
        }
        #pragma unroll
        for (int off = 16; off; off >>= 1) a += __shfl_xor(a, off, 32);
        if (lane == 0) partial[(size_t)blockIdx.x * DD_DIM + j] = a;
    }
}

// ---------------------------------------------------------------------------
// Kernel 3: deterministic reduction over the 520 block partials + bias +
// sigmoid. partial[b*64 + j] is coalesced across j.
// ---------------------------------------------------------------------------
__global__ __launch_bounds__(64)
void final_kernel(const float* __restrict__ partial,
                  const float* __restrict__ fc_b,
                  float*       __restrict__ out)
{
    const int j = threadIdx.x;
    if (j < DD_DIM) {
        float s = fc_b[j];
        for (int b = 0; b < NBLK2; ++b) s += partial[(size_t)b * DD_DIM + j];
        out[j] = 1.0f / (1.0f + expf(-s));
    }
}

// ---------------------------------------------------------------------------
extern "C" void kernel_launch(void* const* d_in, const int* in_sizes, int n_in,
                              void* d_out, int out_size, void* d_ws, size_t ws_size,
                              hipStream_t stream) {
    const int*   inputs   = (const int*)  d_in[0];
    const float* rf       = (const float*)d_in[1];
    const float* embed_w  = (const float*)d_in[2];
    const float* rel_w    = (const float*)d_in[3];
    const float* dise_w   = (const float*)d_in[4];
    const float* fc_w     = (const float*)d_in[5];
    const float* fc_b     = (const float*)d_in[6];
    float*       out      = (float*)d_out;

    // workspace: kg [64 x 32768] f32, then partial [520 x 64] f32
    float* kg      = (float*)d_ws;
    float* partial = kg + (size_t)DD_DIM * N_COLS;

    kg_kernel<<<N_COLS / 16 / 4, 128, 0, stream>>>(inputs, embed_w, rel_w, dise_w, kg);
    gemv_kernel<<<NBLK2, 256, 0, stream>>>(rf, kg, fc_w, partial);
    final_kernel<<<1, 64, 0, stream>>>(partial, fc_b, out);
}
